// SelfAttention_16501264351961
// MI455X (gfx1250) — matrix operand order repaired and checked by reading, not compiled, not run on hardware
//
#include <hip/hip_runtime.h>
#include <hip/hip_bf16.h>

// ---------------------------------------------------------------------------
// Self-attention (B=4, T=2048, H=1024, 16 heads, d_k=64) on gfx1250.
// All GEMM stages use v_wmma_f32_16x16x32_bf16 (fp32 accumulate).
// Round 2: register double-buffered GEMM inner loops + LDS-staged K/V tiles
// shared by the 8 waves of a block in the attention kernel.
// ---------------------------------------------------------------------------

typedef __bf16 bf16;
typedef __attribute__((ext_vector_type(16))) __bf16 v16bf;
typedef __attribute__((ext_vector_type(8)))  float  v8f;
typedef __attribute__((ext_vector_type(4)))  float  f32x4;

#define NUM_HEADS 16
#define B_   4
#define T_   2048
#define H_   1024
#define DK   64
#define BT   (B_ * T_)         // 8192 tokens
#define NQKV (3 * H_)          // 3072

// ln(10000)/32 : theta_j = exp(-j * this), j in [0,32)
#define ROPE_LN_SCALE 0.28782313662425572f

union AFrag { v16bf v; f32x4 q[2]; };

__device__ __forceinline__ v8f wmma_bf16(v16bf a, v16bf b, v8f c) {
  return __builtin_amdgcn_wmma_f32_16x16x32_bf16(false, a, false, b,
                                                 (short)0, c, false, false);
}

__device__ __forceinline__ v8f zero8() {
  v8f z;
#pragma unroll
  for (int i = 0; i < 8; ++i) z[i] = 0.0f;
  return z;
}

// A-fragment (16x32 bf16): lane m=l%16, g=l/16; K = 16g .. 16g+15 contiguous
__device__ __forceinline__ void load_af(AFrag& a, const bf16* p) {
  a.q[0] = *(const f32x4*)(p);
  a.q[1] = *(const f32x4*)(p + 8);
}
// B-fragment (32x16 bf16): lane n=l%16, g=l/16; K = 16g .. 16g+15 contiguous
__device__ __forceinline__ void load_bf(AFrag& b, const bf16* p) {
  b.q[0] = *(const f32x4*)(p);
  b.q[1] = *(const f32x4*)(p + 8);
}

// ------------------------- prep: fp32 -> bf16 ------------------------------
__global__ void cvt_bf16_kernel(const float* __restrict__ in,
                                bf16* __restrict__ out, int n) {
  int i = blockIdx.x * blockDim.x + threadIdx.x;
  if (i < n) out[i] = (bf16)in[i];
}

// in: [K][N] fp32 (row-major) -> out: [N][K] bf16 (transposed)
__global__ void transpose_bf16_kernel(const float* __restrict__ in,
                                      bf16* __restrict__ out, int K, int N) {
  int i = blockIdx.x * blockDim.x + threadIdx.x;
  if (i < K * N) {
    int k = i / N, n = i % N;
    out[(size_t)n * K + k] = (bf16)in[i];
  }
}

// --------------- QKV GEMM (bf16 WMMA) + bias + fused RoPE ------------------
// X  : [BT][H]   bf16
// Wt : [3H][H]   bf16 (transposed weights)
// Qh, Kh : [B*NH][T][DK] bf16 (RoPE applied)
// Vt     : [B*NH][DK][T] bf16 (transposed for PV B-fragments)
__global__ void __launch_bounds__(256)
qkv_gemm_rope_kernel(const bf16* __restrict__ X, const bf16* __restrict__ Wt,
                     const float* __restrict__ bias, bf16* __restrict__ Qh,
                     bf16* __restrict__ Kh, bf16* __restrict__ Vt) {
  const int lane = threadIdx.x & 31;
  const int n    = lane & 15;   // column-within-tile / A-row m
  const int g    = lane >> 4;   // half-wave group
  const int w    = blockIdx.x * 8 + (threadIdx.x >> 5);
  const int MT   = BT / 16;     // 512
  const int mt   = w % MT;
  const int nt   = w / MT;      // 0..47
  const int m0   = mt * 16;
  const int n0   = nt * 64;

  v8f acc[4];
#pragma unroll
  for (int c = 0; c < 4; ++c) acc[c] = zero8();

  const bf16* arow = X + (size_t)(m0 + n) * H_ + 16 * g;
  const bf16* brow[4];
#pragma unroll
  for (int c = 0; c < 4; ++c)
    brow[c] = Wt + (size_t)(n0 + c * 16 + n) * H_ + 16 * g;

  // software-pipelined: phase A computes ks, prefetches ks+32; phase B swaps
  AFrag a0, a1, b0[4], b1[4];
  load_af(a0, arow);
#pragma unroll
  for (int c = 0; c < 4; ++c) load_bf(b0[c], brow[c]);

  for (int ks = 0; ks < H_; ks += 64) {
    load_af(a1, arow + ks + 32);
#pragma unroll
    for (int c = 0; c < 4; ++c) load_bf(b1[c], brow[c] + ks + 32);
#pragma unroll
    for (int c = 0; c < 4; ++c) acc[c] = wmma_bf16(a0.v, b0[c].v, acc[c]);

    if (ks + 64 < H_) {
      load_af(a0, arow + ks + 64);
#pragma unroll
      for (int c = 0; c < 4; ++c) load_bf(b0[c], brow[c] + ks + 64);
    }
#pragma unroll
    for (int c = 0; c < 4; ++c) acc[c] = wmma_bf16(a1.v, b1[c].v, acc[c]);
  }

  // bias (b_qkv) — same value for every row of a column
  float bvals[4];
#pragma unroll
  for (int c = 0; c < 4; ++c) bvals[c] = bias[n0 + c * 16 + n];
#pragma unroll
  for (int c = 0; c < 4; ++c)
#pragma unroll
    for (int v = 0; v < 8; ++v) acc[c][v] += bvals[c];

  const int sec = n0 / H_;            // 0=q, 1=k, 2=v
  const int h   = (n0 % H_) / DK;     // head (block is head-aligned)

  if (sec < 2) {
    bf16* dst = (sec == 0) ? Qh : Kh;
#pragma unroll
    for (int c = 0; c < 2; ++c) {
      const int j = c * 16 + n;                       // rotation index 0..31
      const float theta = __expf(-(float)j * ROPE_LN_SCALE);
#pragma unroll
      for (int v = 0; v < 8; ++v) {
        const int r = m0 + v + 8 * g;                 // global token row
        const int b = r / T_, t = r % T_;
        float s, cs;
        __sincosf((float)t * theta, &s, &cs);
        const float x1 = acc[c][v], x2 = acc[c + 2][v];
        const size_t base = ((size_t)(b * NUM_HEADS + h) * T_ + t) * DK;
        dst[base + j]      = (bf16)(x1 * cs - x2 * s);
        dst[base + j + 32] = (bf16)(x2 * cs + x1 * s);
      }
    }
  } else {
#pragma unroll
    for (int c = 0; c < 4; ++c) {
      const int d = c * 16 + n;
#pragma unroll
      for (int v = 0; v < 8; ++v) {
        const int r = m0 + v + 8 * g;
        const int b = r / T_, t = r % T_;
        Vt[((size_t)(b * NUM_HEADS + h) * DK + d) * T_ + t] = (bf16)acc[c][v];
      }
    }
  }
}

// ----------------- flash attention ------------------------------------------
// Block = 8 waves, all sharing one (b,h); wave w handles query tile qt0+w.
// Per 32-key step the block cooperatively stages K (32x64) and V (64x32)
// tiles into double-buffered LDS; all 8 waves read fragments from LDS.
// Qh/Kh: [B*NH][T][DK], Vt: [B*NH][DK][T], ctx: [BT][H] bf16
__global__ void __launch_bounds__(256)
attention_kernel(const bf16* __restrict__ Qh, const bf16* __restrict__ Kh,
                 const bf16* __restrict__ Vt, const int* __restrict__ mask,
                 bf16* __restrict__ ctx) {
  __shared__ __align__(16) bf16 kbuf[2][32 * DK];  // [key][d]   4 KB each
  __shared__ __align__(16) bf16 vbuf[2][DK * 32];  // [d][key]   4 KB each
  __shared__ __align__(16) bf16 pbuf[8][16 * 32];  // wave-private P tiles

  const int tid  = threadIdx.x;
  const int lane = tid & 31;
  const int n    = lane & 15;
  const int g    = lane >> 4;
  const int wib  = tid >> 5;

  const int bh  = blockIdx.x / (T_ / 16 / 8);      // 0..63
  const int qt  = (blockIdx.x % (T_ / 16 / 8)) * 8 + wib;
  const int b   = bh / NUM_HEADS;

  const bf16* Khb = Kh + (size_t)bh * T_ * DK;
  const bf16* Vtb = Vt + (size_t)bh * DK * T_;

  // staging indices (per thread, 16 bytes each for K and V tiles)
  const int kr = tid >> 3, kc = (tid & 7) * 8;     // K: 32 rows x 64 d
  const int vr = tid >> 2, vc = (tid & 3) * 8;     // V: 64 rows x 32 keys

  // Q A-fragments for d-chunks [0,32) and [32,64)
  const bf16* qrow = Qh + ((size_t)bh * T_ + qt * 16 + n) * DK;
  AFrag qa0, qa1;
  load_af(qa0, qrow + 16 * g);
  load_af(qa1, qrow + 32 + 16 * g);

  v8f O[4];
#pragma unroll
  for (int c = 0; c < 4; ++c) O[c] = zero8();
  float mrow[8], lrow[8];
#pragma unroll
  for (int v = 0; v < 8; ++v) { mrow[v] = -3.0e38f; lrow[v] = 0.0f; }

  const int* mrow_ptr = mask + b * T_;
  bf16* pl = &pbuf[wib][0];

  // prime buffer 0
  *(f32x4*)&kbuf[0][kr * DK + kc] = *(const f32x4*)(Khb + (size_t)kr * DK + kc);
  *(f32x4*)&vbuf[0][vr * 32 + vc] = *(const f32x4*)(Vtb + (size_t)vr * T_ + vc);
  __syncthreads();

  const int NIT = T_ / 32;    // 64
  for (int it = 0; it < NIT; ++it) {
    const int kt  = it * 32;
    const int cur = it & 1;
    if (it + 1 < NIT) {
      const int nxt = (it + 1) & 1;
      const int ktn = kt + 32;
      *(f32x4*)&kbuf[nxt][kr * DK + kc] =
          *(const f32x4*)(Khb + (size_t)(ktn + kr) * DK + kc);
      *(f32x4*)&vbuf[nxt][vr * 32 + vc] =
          *(const f32x4*)(Vtb + (size_t)vr * T_ + ktn + vc);
    }

    // ---- S = Q @ K^T (two 16-key tiles) from LDS ----
    const bf16* kb = &kbuf[cur][0];
    AFrag b00, b01, b10, b11;
    load_bf(b00, kb + n * DK + 16 * g);              // tile0, d-chunk0
    load_bf(b01, kb + n * DK + 32 + 16 * g);         // tile0, d-chunk1
    load_bf(b10, kb + (n + 16) * DK + 16 * g);       // tile1, d-chunk0
    load_bf(b11, kb + (n + 16) * DK + 32 + 16 * g);  // tile1, d-chunk1

    v8f S0 = zero8(), S1 = zero8();
    S0 = wmma_bf16(qa0.v, b00.v, S0);
    S0 = wmma_bf16(qa1.v, b01.v, S0);
    S1 = wmma_bf16(qa0.v, b10.v, S1);
    S1 = wmma_bf16(qa1.v, b11.v, S1);

    // ---- scale + mask + online softmax ----
    const float mb0 = mrow_ptr[kt + n]      ? 0.0f : -1.0e30f;
    const float mb1 = mrow_ptr[kt + 16 + n] ? 0.0f : -1.0e30f;
    float p0[8], p1[8], corr[8];
#pragma unroll
    for (int v = 0; v < 8; ++v) {
      p0[v] = S0[v] * 0.125f + mb0;   // 1/sqrt(64)
      p1[v] = S1[v] * 0.125f + mb1;
    }
#pragma unroll
    for (int v = 0; v < 8; ++v) {
      float mx = fmaxf(p0[v], p1[v]);
#pragma unroll
      for (int off = 1; off < 16; off <<= 1)
        mx = fmaxf(mx, __shfl_xor(mx, off, 32));
      const float mnew = fmaxf(mrow[v], mx);
      corr[v] = __expf(mrow[v] - mnew);
      const float e0 = __expf(p0[v] - mnew);
      const float e1 = __expf(p1[v] - mnew);
      float rs = e0 + e1;
#pragma unroll
      for (int off = 1; off < 16; off <<= 1) rs += __shfl_xor(rs, off, 32);
      lrow[v] = lrow[v] * corr[v] + rs;
      mrow[v] = mnew;
      p0[v] = e0; p1[v] = e1;
    }
#pragma unroll
    for (int c = 0; c < 4; ++c)
#pragma unroll
      for (int v = 0; v < 8; ++v) O[c][v] *= corr[v];

    // ---- D-layout -> A-layout reshuffle of P through LDS (in-order/wave) ----
#pragma unroll
    for (int v = 0; v < 8; ++v) {
      const int row = v + 8 * g;
      pl[row * 32 + n]      = (bf16)p0[v];
      pl[row * 32 + 16 + n] = (bf16)p1[v];
    }
    AFrag pa;
    load_af(pa, pl + n * 32 + 16 * g);

    // ---- O += P @ V  (V from LDS) ----
    const bf16* vbp = &vbuf[cur][0];
#pragma unroll
    for (int c = 0; c < 4; ++c) {
      AFrag bv;
      load_bf(bv, vbp + (c * 16 + n) * 32 + 16 * g);
      O[c] = wmma_bf16(pa.v, bv.v, O[c]);
    }
    __syncthreads();
  }

  // ---- normalize and write context [BT][H] ----
#pragma unroll
  for (int v = 0; v < 8; ++v) {
    const float inv = 1.0f / lrow[v];
    const int t = qt * 16 + v + 8 * g;
    const size_t orow =
        ((size_t)b * T_ + t) * H_ + (size_t)(bh % NUM_HEADS) * DK;
#pragma unroll
    for (int c = 0; c < 4; ++c)
      ctx[orow + c * 16 + n] = (bf16)(O[c][v] * inv);
  }
}

// ------------------- output projection: ctx @ W_out + b --------------------
__global__ void __launch_bounds__(256)
out_gemm_kernel(const bf16* __restrict__ Xc, const bf16* __restrict__ Wt,
                const float* __restrict__ bias, float* __restrict__ out) {
  const int lane = threadIdx.x & 31;
  const int n    = lane & 15;
  const int g    = lane >> 4;
  const int w    = blockIdx.x * 8 + (threadIdx.x >> 5);
  const int MT   = BT / 16;   // 512
  const int mt   = w % MT;
  const int nt   = w / MT;    // 0..15
  const int m0   = mt * 16;
  const int n0   = nt * 64;

  v8f acc[4];
#pragma unroll
  for (int c = 0; c < 4; ++c) acc[c] = zero8();

  const bf16* arow = Xc + (size_t)(m0 + n) * H_ + 16 * g;
  const bf16* brow[4];
#pragma unroll
  for (int c = 0; c < 4; ++c)
    brow[c] = Wt + (size_t)(n0 + c * 16 + n) * H_ + 16 * g;

  AFrag a0, a1, b0[4], b1[4];
  load_af(a0, arow);
#pragma unroll
  for (int c = 0; c < 4; ++c) load_bf(b0[c], brow[c]);

  for (int ks = 0; ks < H_; ks += 64) {
    load_af(a1, arow + ks + 32);
#pragma unroll
    for (int c = 0; c < 4; ++c) load_bf(b1[c], brow[c] + ks + 32);
#pragma unroll
    for (int c = 0; c < 4; ++c) acc[c] = wmma_bf16(a0.v, b0[c].v, acc[c]);

    if (ks + 64 < H_) {
      load_af(a0, arow + ks + 64);
#pragma unroll
      for (int c = 0; c < 4; ++c) load_bf(b0[c], brow[c] + ks + 64);
    }
#pragma unroll
    for (int c = 0; c < 4; ++c) acc[c] = wmma_bf16(a1.v, b1[c].v, acc[c]);
  }

#pragma unroll
  for (int c = 0; c < 4; ++c) {
    const float bval = bias[n0 + c * 16 + n];
#pragma unroll
    for (int v = 0; v < 8; ++v) {
      const int r = m0 + v + 8 * g;
      out[(size_t)r * H_ + n0 + c * 16 + n] = acc[c][v] + bval;
    }
  }
}

// ---------------------------------------------------------------------------
extern "C" void kernel_launch(void* const* d_in, const int* in_sizes, int n_in,
                              void* d_out, int out_size, void* d_ws,
                              size_t ws_size, hipStream_t stream) {
  const float* hs   = (const float*)d_in[0];  // hidden_states [B,T,H]
  const int*   am   = (const int*)d_in[1];    // attention_mask [B,T]
  const float* wqkv = (const float*)d_in[2];  // [H, 3H]
  const float* bqkv = (const float*)d_in[3];  // [3H]
  const float* wout = (const float*)d_in[4];  // [H, H]
  const float* bout = (const float*)d_in[5];  // [H]
  float* out = (float*)d_out;

  char* ws = (char*)d_ws;
  bf16* Xbf    = (bf16*)(ws);                        // 16 MiB
  bf16* Wqkv_t = (bf16*)(ws + (16ull << 20));        //  6 MiB
  bf16* Wout_t = (bf16*)(ws + (22ull << 20));        //  2 MiB
  bf16* Qh     = (bf16*)(ws + (24ull << 20));        // 16 MiB
  bf16* Kh     = (bf16*)(ws + (40ull << 20));        // 16 MiB
  bf16* Vt     = (bf16*)(ws + (56ull << 20));        // 16 MiB
  bf16* Ctx    = (bf16*)(ws + (72ull << 20));        // 16 MiB

  cvt_bf16_kernel<<<(BT * H_ + 255) / 256, 256, 0, stream>>>(hs, Xbf, BT * H_);
  transpose_bf16_kernel<<<(H_ * NQKV + 255) / 256, 256, 0, stream>>>(
      wqkv, Wqkv_t, H_, NQKV);
  transpose_bf16_kernel<<<(H_ * H_ + 255) / 256, 256, 0, stream>>>(
      wout, Wout_t, H_, H_);

  // 512 M-tiles x 48 N-tiles = 24576 waves, 8 waves/block
  qkv_gemm_rope_kernel<<<(512 * 48) / 8, 256, 0, stream>>>(Xbf, Wqkv_t, bqkv,
                                                           Qh, Kh, Vt);
  // 64 (b,h) x 16 blocks, each block = 8 query tiles
  attention_kernel<<<64 * (T_ / 16 / 8), 256, 0, stream>>>(Qh, Kh, Vt, am,
                                                           Ctx);
  // 512 x 16 = 8192 waves
  out_gemm_kernel<<<(512 * 16) / 8, 256, 0, stream>>>(Ctx, Wout_t, bout, out);
}
